// self_LSTM_sparse_attn_48172353192715
// MI455X (gfx1250) — matrix operand order repaired and checked by reading, not compiled, not run on hardware
//
#include <hip/hip_runtime.h>
#include <hip/hip_bf16.h>

typedef __attribute__((ext_vector_type(16))) __bf16 v16bf;
typedef __attribute__((ext_vector_type(8)))  __bf16 v8bf;
typedef __attribute__((ext_vector_type(8)))  float  v8f;

#define B_   64
#define S_   512
#define D_   256
#define H_   512
#define C_   64
#define G4H  2048        // 4*H
#define KC   768         // D + H

// ---- workspace layout (bytes, all offsets 256B-aligned) ----
#define WS_WCB   0u          // Wc bf16   [2048][768]  : 3,145,728
#define WS_WFCB  3145728u    // Wfc bf16  [64][1024]   :   131,072
#define WS_BIASC 3276800u    // b_ih+b_hh f32 [2048]   :     8,192
#define WS_GATES 3284992u    // gates f32 [64][2048]   :   524,288
#define WS_C     3809280u    // c f32     [64][512]    :   131,072
#define WS_HBF   3940352u    // h bf16    [64][512]    :    65,536
#define WS_MT    4005888u    // m_t f32   [64][512]    :   131,072
#define WS_MPREV 4136960u    // m_prev bf16 [64][512]  :    65,536
#define WS_TV    4202496u    // top vals f32 [64][16]  :     4,096
#define WS_TX    4206592u    // top idx  i32 [64][16]  :     4,096
#define WS_CNT   4210688u    // counts   i32 [64]      :       256
#define WS_HSEQ  4210944u    // hseq bf16 [512][64][512]: 33,554,432  (doubles as "mem" slots 1..512)
// total ~37.8 MB

__device__ __forceinline__ float sigf(float x) { return 1.0f / (1.0f + __expf(-x)); }

__device__ __forceinline__ v8f wmma_bf16(v16bf a, v16bf b, v8f c) {
  // D = A(16x32 bf16) * B(32x16 bf16) + C(16x16 f32)
  return __builtin_amdgcn_wmma_f32_16x16x32_bf16(false, a, false, b, (short)0, c, false, false);
}

__device__ __forceinline__ v16bf pack_a(v8bf lo, v8bf hi) {
  v16bf a;
#pragma unroll
  for (int e = 0; e < 8; ++e) { a[e] = lo[e]; a[e + 8] = hi[e]; }
  return a;
}

__device__ __forceinline__ v8bf cvt8(v8f f) {
  v8bf r;
#pragma unroll
  for (int e = 0; e < 8; ++e) r[e] = (__bf16)f[e];
  return r;
}

// ---------------- one-time weight prep ----------------
__global__ void prep_kernel(const float* __restrict__ Wih, const float* __restrict__ Whh,
                            const float* __restrict__ bih, const float* __restrict__ bhh,
                            const float* __restrict__ Wfc,
                            __bf16* __restrict__ Wcb, __bf16* __restrict__ Wfcb,
                            float* __restrict__ biasc) {
  int idx = blockIdx.x * 256 + threadIdx.x;
  if (idx < G4H * KC) {
    int n = idx / KC, k = idx - n * KC;
    float v = (k < D_) ? Wih[n * D_ + k] : Whh[n * H_ + (k - D_)];
    Wcb[idx] = (__bf16)v;
  }
  if (idx < G4H) biasc[idx] = bih[idx] + bhh[idx];
  if (idx < C_ * 2 * H_) Wfcb[idx] = (__bf16)Wfc[idx];
}

// ---------------- per-call state init ----------------
__global__ void init_kernel(float* __restrict__ c, float* __restrict__ mt,
                            __bf16* __restrict__ hbf, __bf16* __restrict__ mprev,
                            float* __restrict__ tv, int* __restrict__ tx,
                            int* __restrict__ cnt) {
  int idx = blockIdx.x * 256 + threadIdx.x;
  if (idx < B_ * H_) {
    c[idx] = 0.0f; mt[idx] = 0.0f;
    hbf[idx] = (__bf16)0.0f; mprev[idx] = (__bf16)0.0f;
  }
  if (idx < B_) {
    cnt[idx] = 1;                       // slot 0 (zero memory row, score 0) is valid
#pragma unroll
    for (int t = 0; t < 16; ++t) {
      tv[idx * 16 + t] = (t == 0) ? 0.0f : -1.0e30f;
      tx[idx * 16 + t] = 0;
    }
  }
}

// ---------------- per-step WMMA GEMMs ----------------
// blocks 0..63  (512 waves): gates(i) = [x_i | h] @ Wc^T + bias      (64 x 2048, K=768)
// blocks 64..65 ( 16 waves): out(i-1) = [h | m_prev] @ Wfc^T + b_fc  (64 x   64, K=1024)
__global__ void step_gemm(const float* __restrict__ x, const __bf16* __restrict__ Wcb,
                          const float* __restrict__ biasc, const __bf16* __restrict__ hbf,
                          const __bf16* __restrict__ mprev, const __bf16* __restrict__ Wfcb,
                          const float* __restrict__ bfc, float* __restrict__ gates,
                          float* __restrict__ out, int i) {
  const int wave = threadIdx.x >> 5;
  const int lane = threadIdx.x & 31;
  const int lm = lane & 15;        // row/col within tile
  const int cs = lane >> 4;        // K-chunk half select

  if (blockIdx.x < 64) {
    if (i >= S_) return;           // wave-uniform: EXEC stays all-ones
    const int t = blockIdx.x * 8 + wave;    // 0..511
    const int m0 = (t & 3) * 16;            // batch tile
    const int n0 = (t >> 2) * 16;           // gate-column tile
    const int rb = m0 + lm;
    const __bf16* wrow = Wcb + (size_t)(n0 + lm) * KC;

    v8f acc;
    const float bia = biasc[n0 + lm];
#pragma unroll
    for (int e = 0; e < 8; ++e) acc[e] = bia;

    // K = 0..255 : x_t (f32 -> bf16 on the fly);  x layout (B,S,D)
    const float* xrow = x + ((size_t)rb * S_ + (size_t)i) * D_;
#pragma unroll
    for (int kk = 0; kk < D_; kk += 32) {
      const int ka = kk + 8 * cs;
      v8f f0 = *(const v8f*)(xrow + ka);
      v8f f1 = *(const v8f*)(xrow + ka + 16);
      v16bf a = pack_a(cvt8(f0), cvt8(f1));
      v16bf bm = pack_a(*(const v8bf*)(wrow + ka), *(const v8bf*)(wrow + ka + 16));
      __builtin_prefetch(wrow + kk + 64, 0, 1);
      acc = wmma_bf16(a, bm, acc);
    }
    // K = 256..767 : recurrent h (bf16)
    const __bf16* hrowp = hbf + (size_t)rb * H_;
#pragma unroll 4
    for (int kk = 0; kk < H_; kk += 32) {
      const int ka = kk + 8 * cs;
      v8bf l0 = *(const v8bf*)(hrowp + ka);
      v8bf l1 = *(const v8bf*)(hrowp + ka + 16);
      v16bf a = pack_a(l0, l1);
      v16bf bm = pack_a(*(const v8bf*)(wrow + D_ + ka), *(const v8bf*)(wrow + D_ + ka + 16));
      acc = wmma_bf16(a, bm, acc);
    }
#pragma unroll
    for (int v = 0; v < 8; ++v)
      gates[(size_t)(m0 + v + 8 * cs) * G4H + (n0 + lm)] = acc[v];
  } else {
    if (i == 0) return;            // wave-uniform
    const int p = i - 1;
    const int ww = (blockIdx.x - 64) * 8 + wave;   // 0..15
    const int m0 = (ww & 3) * 16;
    const int n0 = (ww >> 2) * 16;
    const int rb = m0 + lm;
    const __bf16* wrow = Wfcb + (size_t)(n0 + lm) * (2 * H_);

    v8f acc;
    const float bia = bfc[n0 + lm];
#pragma unroll
    for (int e = 0; e < 8; ++e) acc[e] = bia;

    const __bf16* hrowp = hbf + (size_t)rb * H_;
#pragma unroll 4
    for (int kk = 0; kk < H_; kk += 32) {
      const int ka = kk + 8 * cs;
      v16bf a = pack_a(*(const v8bf*)(hrowp + ka), *(const v8bf*)(hrowp + ka + 16));
      v16bf bm = pack_a(*(const v8bf*)(wrow + ka), *(const v8bf*)(wrow + ka + 16));
      acc = wmma_bf16(a, bm, acc);
    }
    const __bf16* mrowp = mprev + (size_t)rb * H_;
#pragma unroll 4
    for (int kk = 0; kk < H_; kk += 32) {
      const int ka = kk + 8 * cs;
      v16bf a = pack_a(*(const v8bf*)(mrowp + ka), *(const v8bf*)(mrowp + ka + 16));
      v16bf bm = pack_a(*(const v8bf*)(wrow + H_ + ka), *(const v8bf*)(wrow + H_ + ka + 16));
      acc = wmma_bf16(a, bm, acc);
    }
    // out (B,S,C): row = batch, col = class, at step p
#pragma unroll
    for (int v = 0; v < 8; ++v)
      out[(size_t)(m0 + v + 8 * cs) * (S_ * C_) + (size_t)p * C_ + (n0 + lm)] = acc[v];
  }
}

// ---------------- per-step cell + incremental sparse attention ----------------
// one block per batch. Computes LSTM pointwise update, h_t = h_raw + m_t^{(i)},
// writes mem slot i+1, caches its score tanh(h_t)·w_m, updates running top-10,
// and precomputes m_t^{(i+1)} (weighted sum of <=10 memory rows).
__global__ void step_cell(const float* __restrict__ gates, float* __restrict__ c,
                          float* __restrict__ mt, __bf16* __restrict__ hbf,
                          __bf16* __restrict__ mprev, __bf16* __restrict__ hseq,
                          const float* __restrict__ w_m, float* __restrict__ tvg,
                          int* __restrict__ txg, int* __restrict__ cntg, int i) {
  __shared__ float red[256];
  __shared__ float hrow[H_];
  __shared__ float lw[10];
  __shared__ int   lidx[10];
  __shared__ int   lcnt;

  const int b = blockIdx.x;
  const int tid = threadIdx.x;
  const float* g = gates + (size_t)b * G4H;

  float ht_loc[2];
  float partial = 0.0f;
#pragma unroll
  for (int u = 0; u < 2; ++u) {
    const int j = tid + u * 256;
    const float ig = g[j];
    const float fg = g[H_ + j];
    const float gg = g[2 * H_ + j];
    const float og = g[3 * H_ + j];
    const float cp = c[(size_t)b * H_ + j];
    const float cn = sigf(fg) * cp + sigf(ig) * tanhf(gg);
    const float hr = sigf(og) * tanhf(cn);
    const float mtc = mt[(size_t)b * H_ + j];
    const float ht = hr + mtc;                 // h_t = h_raw + m_t^{(i)}
    ht_loc[u] = ht;
    c[(size_t)b * H_ + j] = cn;
    hbf[(size_t)b * H_ + j] = (__bf16)ht;                       // recurrent input for step i+1
    hseq[((size_t)i * B_ + b) * H_ + j] = (__bf16)ht;           // memory slot i+1
    mprev[(size_t)b * H_ + j] = (__bf16)mtc;                    // m_seq[i] for output proj
    hrow[j] = ht;
    partial += tanhf(ht) * w_m[j];             // cached slot score (w_h term cancels exactly)
  }
  red[tid] = partial;
  __syncthreads();
#pragma unroll
  for (int s = 128; s > 0; s >>= 1) {
    if (tid < s) red[tid] += red[tid + s];
    __syncthreads();
  }

  if (tid == 0) {
    float tvl[10]; int txl[10];
    float* tvp = tvg + b * 16; int* txp = txg + b * 16;
    int cnt = cntg[b];
    for (int t = 0; t < 10; ++t) { if (t < cnt) { tvl[t] = tvp[t]; txl[t] = txp[t]; } }
    const float sc = red[0];
    const int sl = i + 1;                      // slot written this step, valid from step i+1
    if (cnt < 10) { tvl[cnt] = sc; txl[cnt] = sl; cnt++; }
    else if (sc > tvl[9]) { tvl[9] = sc; txl[9] = sl; }
    for (int t = 9; t > 0; --t) {
      if (t < cnt && tvl[t] > tvl[t - 1]) {
        float fv = tvl[t]; tvl[t] = tvl[t - 1]; tvl[t - 1] = fv;
        int iv = txl[t]; txl[t] = txl[t - 1]; txl[t - 1] = iv;
      }
    }
    for (int t = 0; t < 10; ++t) { if (t < cnt) { tvp[t] = tvl[t]; txp[t] = txl[t]; } }
    cntg[b] = cnt;
    // count<=10: threshold = min(valid); count>10: 10th largest.  Both = tvl[cnt-1].
    const float thresh = tvl[cnt - 1] + 1e-7f;
    float wv[10]; float wsum = 0.0f;
    for (int t = 0; t < 10; ++t) {
      float w = 0.0f;
      if (t < cnt) { w = tvl[t] - thresh; w = (w > 0.0f) ? w : 0.0f; }
      wv[t] = w; wsum += w;
    }
    const float inv = 1.0f / (wsum + 1e-7f);
    for (int t = 0; t < 10; ++t) { lw[t] = wv[t] * inv; lidx[t] = (t < cnt) ? txl[t] : 0; }
    lcnt = cnt;
  }
  __syncthreads();

  // m_t^{(i+1)} = sum over top-k of aw * mem_row   (<=10 rows of 512)
  const int nsel = lcnt;
#pragma unroll
  for (int u = 0; u < 2; ++u) {
    const int j = tid + u * 256;
    float acc = 0.0f;
    for (int t = 0; t < nsel; ++t) {
      const float w = lw[t];
      if (w != 0.0f) {
        const int slot = lidx[t];
        float v;
        if (slot == i + 1)      v = hrow[j];                                 // just written (LDS)
        else if (slot == 0)     v = 0.0f;                                    // zero init row
        else                    v = (float)hseq[((size_t)(slot - 1) * B_ + b) * H_ + j];
        acc += w * v;
      }
    }
    mt[(size_t)b * H_ + j] = acc;
  }
  (void)ht_loc;
}

extern "C" void kernel_launch(void* const* d_in, const int* in_sizes, int n_in,
                              void* d_out, int out_size, void* d_ws, size_t ws_size,
                              hipStream_t stream) {
  const float* x   = (const float*)d_in[0];
  const float* Wih = (const float*)d_in[1];
  const float* Whh = (const float*)d_in[2];
  const float* bih = (const float*)d_in[3];
  const float* bhh = (const float*)d_in[4];
  const float* wt  = (const float*)d_in[5];   // (2H,1): [w_h | w_m]; w_h cancels analytically
  const float* Wfc = (const float*)d_in[6];
  const float* bfc = (const float*)d_in[7];

  char* ws = (char*)d_ws;
  __bf16* Wcb   = (__bf16*)(ws + WS_WCB);
  __bf16* Wfcb  = (__bf16*)(ws + WS_WFCB);
  float*  biasc = (float*)(ws + WS_BIASC);
  float*  gates = (float*)(ws + WS_GATES);
  float*  cbuf  = (float*)(ws + WS_C);
  __bf16* hbf   = (__bf16*)(ws + WS_HBF);
  float*  mt    = (float*)(ws + WS_MT);
  __bf16* mprev = (__bf16*)(ws + WS_MPREV);
  float*  tv    = (float*)(ws + WS_TV);
  int*    tx    = (int*)(ws + WS_TX);
  int*    cnt   = (int*)(ws + WS_CNT);
  __bf16* hseq  = (__bf16*)(ws + WS_HSEQ);

  prep_kernel<<<(G4H * KC + 255) / 256, 256, 0, stream>>>(Wih, Whh, bih, bhh, Wfc,
                                                          Wcb, Wfcb, biasc);
  init_kernel<<<(B_ * H_ + 255) / 256, 256, 0, stream>>>(cbuf, mt, hbf, mprev, tv, tx, cnt);

  for (int i = 0; i <= S_; ++i) {
    // i < S : gate GEMM for step i;  i > 0 : output projection for step i-1
    step_gemm<<<66, 256, 0, stream>>>(x, Wcb, biasc, hbf, mprev, Wfcb, bfc,
                                      gates, (float*)d_out, i);
    if (i < S_)
      step_cell<<<B_, 256, 0, stream>>>(gates, cbuf, mt, hbf, mprev, hseq,
                                        wt + H_, tv, tx, cnt, i);
  }
  (void)in_sizes; (void)n_in; (void)out_size; (void)ws_size;
}
